// ClinicalGAT_78769700209021
// MI455X (gfx1250) — compile-verified, matrix operand order checked by reading, not run on hardware
//
#include <hip/hip_runtime.h>
#include <hip/hip_bf16.h>
#include <math.h>

#define NN 10000          // nodes
#define NE 160000         // edges (before self loops)
#define NL (NE + NN)      // edges incl self loops
#define NG 64             // graphs
#define NPAD 10048        // nodes padded to multiple of 64 (GEMM BM)

typedef __attribute__((ext_vector_type(16))) __bf16 v16bf;
typedef __attribute__((ext_vector_type(8)))  __bf16 v8bf;
typedef __attribute__((ext_vector_type(8)))  float  v8f;
typedef __attribute__((ext_vector_type(4)))  unsigned int u32x4;

__device__ __forceinline__ unsigned short f2bf(float f) {
  unsigned int x = __float_as_uint(f);
  unsigned int r = x + 0x7FFFu + ((x >> 16) & 1u);   // round-to-nearest-even
  return (unsigned short)(r >> 16);
}

__device__ __forceinline__ void atomicMaxF(float* addr, float val) {
  unsigned int* ua = (unsigned int*)addr;
  unsigned int old = *ua;
  while (true) {
    float f = __uint_as_float(old);
    if (f >= val) break;
    unsigned int assumed = old;
    old = atomicCAS(ua, assumed, __float_as_uint(val));
    if (old == assumed) break;
  }
}

// ---------------------------------------------------------------------------
// bf16 WMMA GEMM:
//   C[M,N] f32 = A[Mpad,K]bf16 (row-major) @ B, with B pre-transposed as
//   Bt[N,K] bf16 so both LDS tiles are K-contiguous per lane.
// block = 256 threads (8 waves), tile 64x128, K-step 32, wave tile 32x32.
// Software-pipelined; all staging in 16-byte vector ops (b128).
// ---------------------------------------------------------------------------
#define BM 64
#define BN 128
#define BK 32
#define LSTR 48   // ushort row stride in LDS: 96 bytes (multiple of 16 -> b128 aligned)

__global__ __launch_bounds__(256)
void gemm_bf16_wmma(const __bf16* __restrict__ A,
                    const __bf16* __restrict__ Bt,
                    float* __restrict__ C,
                    int Mvalid, int K, int N)
{
  __shared__ __bf16 As[BM * LSTR];
  __shared__ __bf16 Bs[BN * LSTR];

  const int tid  = threadIdx.x;
  const int lane = tid & 31;
  const int wave = tid >> 5;
  const int wm   = (wave & 1) * 32;   // 2 waves along M
  const int wn   = (wave >> 1) * 32;  // 4 waves along N
  const int half = lane >> 4;
  const int r16  = lane & 15;
  const int blockM = blockIdx.x * BM;
  const int blockN = blockIdx.y * BN;
  const unsigned K8 = (unsigned)K >> 3;   // row stride in uint4 (16B) units

  const u32x4* Ag  = (const u32x4*)A;
  const u32x4* Bg  = (const u32x4*)Bt;
  u32x4*       Asv = (u32x4*)As;          // LSTR/8 = 6 uint4 per LDS row
  u32x4*       Bsv = (u32x4*)Bs;

  // A tile: 64 rows x 4 chunks(16B) = 256 -> 1 chunk per thread.
  const int arow = tid >> 2, ach = tid & 3;
  unsigned aoff = (unsigned)(blockM + arow) * K8 + (unsigned)ach;
  const unsigned alds = (unsigned)(arow * 6 + ach);
  // B tile: 128 cols x 4 chunks = 512 -> 2 chunks per thread.
  unsigned boff[2], blds[2];
#pragma unroll
  for (int it = 0; it < 2; ++it) {
    int idx = tid + it * 256;
    int col = idx >> 2, ch = idx & 3;
    boff[it] = (unsigned)(blockN + col) * K8 + (unsigned)ch;
    blds[it] = (unsigned)(col * 6 + ch);
  }

  v8f c00 = {0.f,0.f,0.f,0.f,0.f,0.f,0.f,0.f};
  v8f c01 = c00, c10 = c00, c11 = c00;

  u32x4 ra, rb0, rb1;
  ra  = Ag[aoff];
  rb0 = Bg[boff[0]];
  rb1 = Bg[boff[1]];

  const int a0base = (wm + 0  + r16) * LSTR + half * 8;
  const int a1base = (wm + 16 + r16) * LSTR + half * 8;
  const int b0base = (wn + 0  + r16) * LSTR + half * 8;
  const int b1base = (wn + 16 + r16) * LSTR + half * 8;

  for (int k0 = 0; k0 < K; k0 += BK) {
    __syncthreads();                 // previous compute done reading LDS
    Asv[alds]    = ra;
    Bsv[blds[0]] = rb0;
    Bsv[blds[1]] = rb1;
    __syncthreads();

    if (k0 + BK < K) {               // issue next tile's global loads early
      aoff    += BK / 8;
      boff[0] += BK / 8;
      boff[1] += BK / 8;
      ra  = Ag[aoff];
      rb0 = Bg[boff[0]];
      rb1 = Bg[boff[1]];
    }

    // Fragments: two aligned b128 chunks per lane, concatenated.
    v8bf al0 = *(const v8bf*)&As[a0base];
    v8bf ah0 = *(const v8bf*)&As[a0base + 16];
    v8bf al1 = *(const v8bf*)&As[a1base];
    v8bf ah1 = *(const v8bf*)&As[a1base + 16];
    v8bf bl0 = *(const v8bf*)&Bs[b0base];
    v8bf bh0 = *(const v8bf*)&Bs[b0base + 16];
    v8bf bl1 = *(const v8bf*)&Bs[b1base];
    v8bf bh1 = *(const v8bf*)&Bs[b1base + 16];
    v16bf a0 = __builtin_shufflevector(al0, ah0, 0,1,2,3,4,5,6,7,8,9,10,11,12,13,14,15);
    v16bf a1 = __builtin_shufflevector(al1, ah1, 0,1,2,3,4,5,6,7,8,9,10,11,12,13,14,15);
    v16bf b0 = __builtin_shufflevector(bl0, bh0, 0,1,2,3,4,5,6,7,8,9,10,11,12,13,14,15);
    v16bf b1 = __builtin_shufflevector(bl1, bh1, 0,1,2,3,4,5,6,7,8,9,10,11,12,13,14,15);

    c00 = __builtin_amdgcn_wmma_f32_16x16x32_bf16(false, a0, false, b0, (short)0, c00, false, false);
    c01 = __builtin_amdgcn_wmma_f32_16x16x32_bf16(false, a0, false, b1, (short)0, c01, false, false);
    c10 = __builtin_amdgcn_wmma_f32_16x16x32_bf16(false, a1, false, b0, (short)0, c10, false, false);
    c11 = __builtin_amdgcn_wmma_f32_16x16x32_bf16(false, a1, false, b1, (short)0, c11, false, false);
  }

  // Epilogue per C/D layout: VGPR r -> M=r (lanes 0-15) / M=8+r (lanes 16-31), N=lane&15
#pragma unroll
  for (int rr = 0; rr < 8; ++rr) {
    int m0 = blockM + wm + 0  + rr + half * 8;
    int m1 = blockM + wm + 16 + rr + half * 8;
    int n0 = blockN + wn + 0  + r16;
    int n1 = blockN + wn + 16 + r16;
    if (m0 < Mvalid) {
      C[(size_t)m0 * N + n0] = c00[rr];
      C[(size_t)m0 * N + n1] = c01[rr];
    }
    if (m1 < Mvalid) {
      C[(size_t)m1 * N + n0] = c10[rr];
      C[(size_t)m1 * N + n1] = c11[rr];
    }
  }
}

// ---------------------------------------------------------------------------
// fp32 -> bf16 with zero padding (activations): dst[Mpad,Kpad] from src[M,Kin]
// ---------------------------------------------------------------------------
__global__ void cvt_to_bf16_pad(const float* __restrict__ src, unsigned short* __restrict__ dst,
                                int M, int Kin, int Mpad, int Kpad)
{
  size_t total = (size_t)Mpad * Kpad;
  for (size_t i = blockIdx.x * (size_t)blockDim.x + threadIdx.x; i < total;
       i += (size_t)gridDim.x * blockDim.x) {
    int r = (int)(i / Kpad), c = (int)(i % Kpad);
    dst[i] = (r < M && c < Kin) ? f2bf(src[(size_t)r * Kin + c]) : (unsigned short)0;
  }
}

// fp32 W[Fin,Fout] -> bf16 transposed Wt[Fout,Kpad] (zero-padded K)
__global__ void cvt_w_transpose(const float* __restrict__ W, unsigned short* __restrict__ dst,
                                int Fin, int Fout, int Kpad)
{
  size_t total = (size_t)Fout * Kpad;
  for (size_t i = blockIdx.x * (size_t)blockDim.x + threadIdx.x; i < total;
       i += (size_t)gridDim.x * blockDim.x) {
    int c = (int)(i / Kpad), k = (int)(i % Kpad);
    dst[i] = (k < Fin) ? f2bf(W[(size_t)k * Fout + c]) : (unsigned short)0;
  }
}

__global__ void fill_f32(float* __restrict__ p, size_t n, float v) {
  for (size_t i = blockIdx.x * (size_t)blockDim.x + threadIdx.x; i < n;
       i += (size_t)gridDim.x * blockDim.x) p[i] = v;
}

// ---------------------------------------------------------------------------
// Attention logits: one wave per (node, head); 128-dim dot reduced across 32 lanes
// ---------------------------------------------------------------------------
__global__ void attn_logits(const float* __restrict__ h, const float* __restrict__ a_src,
                            const float* __restrict__ a_dst,
                            float* __restrict__ als, float* __restrict__ ald, int H)
{
  int lane = threadIdx.x & 31;
  int wid  = (int)((blockIdx.x * (size_t)blockDim.x + threadIdx.x) >> 5);
  int nw   = (int)((gridDim.x * (size_t)blockDim.x) >> 5);
  int total = NN * H;
  for (int i = wid; i < total; i += nw) {
    int n = i / H, hh = i % H;
    const float* hp = h + ((size_t)n * H + hh) * 128;
    float s1 = 0.f, s2 = 0.f;
    for (int k = lane; k < 128; k += 32) {
      float v = hp[k];
      s1 += v * a_src[hh * 128 + k];
      s2 += v * a_dst[hh * 128 + k];
    }
    for (int m = 16; m >= 1; m >>= 1) {
      s1 += __shfl_xor(s1, m, 32);
      s2 += __shfl_xor(s2, m, 32);
    }
    if (lane == 0) { als[i] = s1; ald[i] = s2; }
  }
}

__device__ __forceinline__ float leaky(float v) { return v > 0.f ? v : 0.2f * v; }

// Pass 1: per-destination max of leaky-relu logits
__global__ void edge_max_k(const int* __restrict__ srcI, const int* __restrict__ dstI,
                           const float* __restrict__ als, const float* __restrict__ ald,
                           float* __restrict__ m, int H)
{
  long total = (long)NL * H;
  for (long i = blockIdx.x * (long)blockDim.x + threadIdx.x; i < total;
       i += (long)gridDim.x * blockDim.x) {
    int hh = (int)(i % H);
    long e = i / H;
    int s = (e < NE) ? srcI[e] : (int)(e - NE);
    int d = (e < NE) ? dstI[e] : (int)(e - NE);
    float v = leaky(als[(size_t)s * H + hh] + ald[(size_t)d * H + hh]);
    atomicMaxF(&m[(size_t)d * H + hh], v);
  }
}

// Pass 2: exp(logit - max), stash per-edge weight, accumulate per-dst sum
__global__ void edge_expsum_k(const int* __restrict__ srcI, const int* __restrict__ dstI,
                              const float* __restrict__ als, const float* __restrict__ ald,
                              const float* __restrict__ m, float* __restrict__ ew,
                              float* __restrict__ ssum, int H)
{
  long total = (long)NL * H;
  for (long i = blockIdx.x * (long)blockDim.x + threadIdx.x; i < total;
       i += (long)gridDim.x * blockDim.x) {
    int hh = (int)(i % H);
    long e = i / H;
    int s = (e < NE) ? srcI[e] : (int)(e - NE);
    int d = (e < NE) ? dstI[e] : (int)(e - NE);
    float v = leaky(als[(size_t)s * H + hh] + ald[(size_t)d * H + hh]);
    float w = expf(v - m[(size_t)d * H + hh]);
    ew[i] = w;
    atomicAdd(&ssum[(size_t)d * H + hh], w);
  }
}

// Pass 3: out[dst] += (w / sum) * h[src]; each thread owns (edge, head, 4 dims)
__global__ void edge_scatter_k(const int* __restrict__ srcI, const int* __restrict__ dstI,
                               const float* __restrict__ ew, const float* __restrict__ ssum,
                               const float* __restrict__ h, float* __restrict__ out, int H)
{
  long total = (long)NL * H * 32;
  for (long i = blockIdx.x * (long)blockDim.x + threadIdx.x; i < total;
       i += (long)gridDim.x * blockDim.x) {
    int d4 = (int)(i & 31) * 4;
    long eh = i >> 5;
    int hh = (int)(eh % H);
    long e = eh / H;
    int s = (e < NE) ? srcI[e] : (int)(e - NE);
    int d = (e < NE) ? dstI[e] : (int)(e - NE);
    float w = ew[eh] / (ssum[(size_t)d * H + hh] + 1e-16f);
    const float* hp = h + ((size_t)s * H + hh) * 128 + d4;
    float* op = out + ((size_t)d * H + hh) * 128 + d4;
#pragma unroll
    for (int j = 0; j < 4; ++j) atomicAdd(&op[j], w * hp[j]);
  }
}

__global__ void bias_elu_k(float* __restrict__ p, const float* __restrict__ b,
                           size_t rows, int F)
{
  size_t total = rows * F;
  for (size_t i = blockIdx.x * (size_t)blockDim.x + threadIdx.x; i < total;
       i += (size_t)gridDim.x * blockDim.x) {
    float v = p[i] + b[i % F];
    p[i] = v > 0.f ? v : (expf(v) - 1.0f);
  }
}

// Global mean pool per graph
__global__ void pool_acc_k(const float* __restrict__ h, const int* __restrict__ batch,
                           float* __restrict__ sums, float* __restrict__ cnt)
{
  size_t total = (size_t)NN * 128;
  for (size_t i = blockIdx.x * (size_t)blockDim.x + threadIdx.x; i < total;
       i += (size_t)gridDim.x * blockDim.x) {
    int n = (int)(i / 128), k = (int)(i % 128);
    int g = batch[n];
    atomicAdd(&sums[(size_t)g * 128 + k], h[i]);
    if (k == 0) atomicAdd(&cnt[g], 1.0f);
  }
}

__global__ void pool_div_k(const float* __restrict__ sums, const float* __restrict__ cnt,
                           float* __restrict__ g)
{
  size_t total = (size_t)NG * 128;
  for (size_t i = blockIdx.x * (size_t)blockDim.x + threadIdx.x; i < total;
       i += (size_t)gridDim.x * blockDim.x)
    g[i] = sums[i] / fmaxf(cnt[i / 128], 1.0f);
}

// Tiny dense layer for the MLP heads (B=64 rows)
__global__ void dense_small(const float* __restrict__ in, const float* __restrict__ W,
                            const float* __restrict__ b, float* __restrict__ out,
                            int din, int dout, int relu, int ostride, int ooff)
{
  int row = blockIdx.x;
  int c = threadIdx.x;
  if (c < dout) {
    float acc = b[c];
    for (int k = 0; k < din; ++k) acc += in[(size_t)row * din + k] * W[(size_t)k * dout + c];
    if (relu) acc = fmaxf(acc, 0.f);
    out[(size_t)row * ostride + ooff + c] = acc;
  }
}

// ---------------------------------------------------------------------------
extern "C" void kernel_launch(void* const* d_in, const int* in_sizes, int n_in,
                              void* d_out, int out_size, void* d_ws, size_t ws_size,
                              hipStream_t stream)
{
  const float* x     = (const float*)d_in[0];
  const int*   ei    = (const int*)d_in[1];
  const int*   batch = (const int*)d_in[2];
  const float* W1  = (const float*)d_in[3];
  const float* as1 = (const float*)d_in[4];
  const float* ad1 = (const float*)d_in[5];
  const float* b1  = (const float*)d_in[6];
  const float* W2  = (const float*)d_in[7];
  const float* as2 = (const float*)d_in[8];
  const float* ad2 = (const float*)d_in[9];
  const float* b2  = (const float*)d_in[10];
  const float* W3  = (const float*)d_in[11];
  const float* as3 = (const float*)d_in[12];
  const float* ad3 = (const float*)d_in[13];
  const float* b3  = (const float*)d_in[14];

  const int* srcI = ei;
  const int* dstI = ei + NE;

  // bump allocator over d_ws
  char* ws = (char*)d_ws;
  size_t off = 0;
  auto alloc = [&](size_t bytes) -> void* {
    off = (off + 255) & ~(size_t)255;
    void* p = ws + off;
    off += bytes;
    return p;
  };
  unsigned short* Abuf = (unsigned short*)alloc((size_t)NPAD * 1024 * 2);
  unsigned short* Bbuf = (unsigned short*)alloc((size_t)1024 * 1024 * 2);
  float* Hbuf = (float*)alloc((size_t)NPAD * 1024 * 4);
  float* Obuf = (float*)alloc((size_t)NPAD * 1024 * 4);
  float* als  = (float*)alloc((size_t)NN * 8 * 4);
  float* ald  = (float*)alloc((size_t)NN * 8 * 4);
  float* mbuf = (float*)alloc((size_t)NN * 8 * 4);
  float* sbuf = (float*)alloc((size_t)NN * 8 * 4);
  float* ewbf = (float*)alloc((size_t)NL * 8 * 4);
  float* psum = (float*)alloc((size_t)NG * 128 * 4);
  float* pcnt = (float*)alloc((size_t)NG * 4);
  float* gbuf = (float*)alloc((size_t)NG * 128 * 4);
  float* t0   = (float*)alloc((size_t)NG * 64 * 4);
  float* t1   = (float*)alloc((size_t)NG * 64 * 4);
  (void)ws_size; (void)out_size; (void)n_in;

  auto blocks = [](size_t total) -> unsigned int {
    return (unsigned int)((total + 255) / 256);
  };

  auto run_gat = [&](const float* inF, int Fin, int KpadA,
                     const float* W, const float* asrc, const float* adst,
                     const float* bias, int H) {
    int Fout = H * 128;
    // activations -> bf16 (padded); weights -> bf16 transposed [Fout, Kpad]
    cvt_to_bf16_pad<<<blocks((size_t)NPAD * KpadA), 256, 0, stream>>>(
        inF, Abuf, NN, Fin, NPAD, KpadA);
    cvt_w_transpose<<<blocks((size_t)Fout * KpadA), 256, 0, stream>>>(
        W, Bbuf, Fin, Fout, KpadA);
    dim3 gg(NPAD / BM, Fout / BN);
    gemm_bf16_wmma<<<gg, 256, 0, stream>>>((const __bf16*)Abuf, (const __bf16*)Bbuf,
                                           Hbuf, NN, KpadA, Fout);
    // attention logits
    attn_logits<<<blocks((size_t)NN * H * 32), 256, 0, stream>>>(
        Hbuf, asrc, adst, als, ald, H);
    // segment softmax + scatter
    fill_f32<<<blocks((size_t)NN * H), 256, 0, stream>>>(mbuf, (size_t)NN * H, -INFINITY);
    fill_f32<<<blocks((size_t)NN * H), 256, 0, stream>>>(sbuf, (size_t)NN * H, 0.f);
    fill_f32<<<blocks((size_t)NN * Fout), 256, 0, stream>>>(Obuf, (size_t)NN * Fout, 0.f);
    edge_max_k<<<blocks((size_t)NL * H), 256, 0, stream>>>(srcI, dstI, als, ald, mbuf, H);
    edge_expsum_k<<<blocks((size_t)NL * H), 256, 0, stream>>>(
        srcI, dstI, als, ald, mbuf, ewbf, sbuf, H);
    edge_scatter_k<<<blocks((size_t)NL * H * 32), 256, 0, stream>>>(
        srcI, dstI, ewbf, sbuf, Hbuf, Obuf, H);
    bias_elu_k<<<blocks((size_t)NN * Fout), 256, 0, stream>>>(Obuf, bias, NN, Fout);
  };

  run_gat(x,    225,  256,  W1, as1, ad1, b1, 8);  // -> Obuf [NN,1024]
  run_gat(Obuf, 1024, 1024, W2, as2, ad2, b2, 4);  // -> Obuf [NN,512]
  run_gat(Obuf, 512,  512,  W3, as3, ad3, b3, 1);  // -> Obuf [NN,128]

  // global mean pool
  fill_f32<<<blocks((size_t)NG * 128), 256, 0, stream>>>(psum, (size_t)NG * 128, 0.f);
  fill_f32<<<1, 64, 0, stream>>>(pcnt, (size_t)NG, 0.f);
  pool_acc_k<<<blocks((size_t)NN * 128), 256, 0, stream>>>(Obuf, batch, psum, pcnt);
  pool_div_k<<<blocks((size_t)NG * 128), 256, 0, stream>>>(psum, pcnt, gbuf);

  // MLP heads -> d_out [64, 18]
  const int dims_all[8][4] = {
      {128, 32, 2, -1}, {128, 32, 2, -1}, {128, 32, 2, -1}, {128, 32, 2, -1},
      {128, 64, 32, 3}, {128, 64, 32, 3},
      {128, 32, 2, -1}, {128, 32, 2, -1}};
  int idx = 15;
  auto nextArr = [&](int sz) -> const float* {
    while (idx < n_in && in_sizes[idx] != sz) ++idx;   // skips size-1 act flags
    const float* p = (const float*)d_in[idx < n_in ? idx : n_in - 1];
    ++idx;
    return p;
  };
  int col_off = 0;
  for (int hd = 0; hd < 8; ++hd) {
    int nd = (dims_all[hd][3] < 0) ? 3 : 4;
    const float* cur = gbuf;
    for (int l = 0; l + 1 < nd; ++l) {
      int din = dims_all[hd][l], dout = dims_all[hd][l + 1];
      const float* Wl = nextArr(din * dout);
      const float* bl = nextArr(dout);
      bool last = (l + 2 == nd);
      float* outp;
      int ostride, ooff;
      if (last) { outp = (float*)d_out; ostride = 18; ooff = col_off; }
      else      { outp = (l % 2 == 0) ? t0 : t1; ostride = dout; ooff = 0; }
      dense_small<<<NG, 64, 0, stream>>>(cur, Wl, bl, outp, din, dout,
                                         last ? 0 : 1, ostride, ooff);
      cur = outp;
    }
    col_off += dims_all[hd][nd - 1];
  }
}
